// EGNNLayer_52578989638102
// MI455X (gfx1250) — matrix operand order repaired and checked by reading, not compiled, not run on hardware
//
#include <hip/hip_runtime.h>
#include <math.h>

// ---------------- problem constants ----------------
#define HID    128
#define NUM_G  20
#define EFEAT  4
#define IN_E   280
#define IN_EP  288          // K padded to multiple of 32
#define NNODE  20000
#define NEDGE  640000
#define CUTOFF 10.0f

// ---------------- WMMA types ----------------
typedef __bf16        v16bf __attribute__((ext_vector_type(16)));
typedef float         v8f   __attribute__((ext_vector_type(8)));
typedef unsigned int  u32x4 __attribute__((ext_vector_type(4)));

// A-fragment (16x32 bf16, MxK): lane = M (lane&15), lane-half picks K+8.
// per lane: 8 contiguous bf16 at [kb + 8*(lane>>4)] and 8 more at +16.
__device__ inline v16bf load_a_frag(const __bf16* p) {
  union { struct { u32x4 lo, hi; } q; v16bf v; } u;
  u.q.lo = *(const u32x4*)(p);
  u.q.hi = *(const u32x4*)(p + 16);
  return u.v;
}

// B-fragment (32x16 bf16, KxN) from N-major transposed weights:
// lane = N (lane&15), lane-half picks K+8; 8 contiguous bf16 per lane and 8 more at +16.
__device__ inline v16bf load_b_frag(const __bf16* p) {
  union { struct { u32x4 lo, hi; } q; v16bf v; } u;
  u.q.lo = *(const u32x4*)(p);
  u.q.hi = *(const u32x4*)(p + 16);
  return u.v;
}

__device__ inline float siluf(float v)    { return v / (1.0f + __expf(-v)); }
__device__ inline float sigmoidf_(float v){ return 1.0f / (1.0f + __expf(-v)); }

// One wave computes a 16(M) x 64(N) tile: 4 WMMA n-tiles, A reused across them.
// As: LDS activations [rows][strideA] bf16 ; Bt: global transposed weights [n][strideB] bf16.
__device__ inline void wmma_gemm(const __bf16* As, int strideA,
                                 const __bf16* Bt, int strideB,
                                 int ksteps, int lane, int mbase, int nb0,
                                 v8f acc[4]) {
  const int mrow = mbase + (lane & 15);
  const int ka   = 8  * (lane >> 4);
  const int kb16 = 16 * (lane >> 4);
  const __bf16* arow = As + mrow * strideA;
#pragma unroll
  for (int t = 0; t < 4; ++t) acc[t] = v8f{0.f,0.f,0.f,0.f,0.f,0.f,0.f,0.f};
  for (int ks = 0; ks < ksteps; ++ks) {
    const int kb = ks * 32;
    v16bf a = load_a_frag(arow + kb + ka);
#pragma unroll
    for (int t = 0; t < 4; ++t) {
      const int n = nb0 + 16 * t + (lane & 15);
      v16bf b = load_b_frag(Bt + n * strideB + kb + ka);
      acc[t] = __builtin_amdgcn_wmma_f32_16x16x32_bf16(
          false, a, false, b, (short)0, acc[t], false, false);
    }
  }
}

// bias + SiLU epilogue -> bf16 LDS tile [row][stride]
__device__ inline void store_silu(const v8f acc[4], const float* bias,
                                  __bf16* out, int stride,
                                  int lane, int mbase, int nb0) {
#pragma unroll
  for (int t = 0; t < 4; ++t) {
    const int n = nb0 + 16 * t + (lane & 15);
    const float b = bias[n];
#pragma unroll
    for (int r = 0; r < 8; ++r) {
      const int row = mbase + r + 8 * (lane >> 4);
      float v = acc[t][r] + b;
      out[row * stride + n] = (__bf16)siluf(v);
    }
  }
}

// ---------------- kernel 0: weight transpose/convert to bf16 ----------------
#define SZ_E1 (HID * IN_EP)     // 36864
#define SZ_SQ (HID * HID)       // 16384
#define SZ_N1 (HID * 256)       // 32768
#define PREP_TOTAL (SZ_E1 + 3 * SZ_SQ + SZ_N1)   // 118784

__global__ void prep_kernel(const float* __restrict__ W_e1, const float* __restrict__ W_e2,
                            const float* __restrict__ W_x1, const float* __restrict__ W_n1,
                            const float* __restrict__ W_n2,
                            __bf16* __restrict__ Wt_e1, __bf16* __restrict__ Wt_e2,
                            __bf16* __restrict__ Wt_x1, __bf16* __restrict__ Wt_n1,
                            __bf16* __restrict__ Wt_n2) {
  int i = blockIdx.x * blockDim.x + threadIdx.x;
  if (i < SZ_E1) {
    int n = i / IN_EP, k = i % IN_EP;
    Wt_e1[i] = (__bf16)((k < IN_E) ? W_e1[k * HID + n] : 0.0f);
  } else if (i < SZ_E1 + SZ_SQ) {
    int j = i - SZ_E1; int n = j >> 7, k = j & 127;
    Wt_e2[j] = (__bf16)W_e2[k * HID + n];
  } else if (i < SZ_E1 + 2 * SZ_SQ) {
    int j = i - SZ_E1 - SZ_SQ; int n = j >> 7, k = j & 127;
    Wt_x1[j] = (__bf16)W_x1[k * HID + n];
  } else if (i < SZ_E1 + 2 * SZ_SQ + SZ_N1) {
    int j = i - SZ_E1 - 2 * SZ_SQ; int n = j >> 8, k = j & 255;
    Wt_n1[j] = (__bf16)W_n1[k * HID + n];
  } else if (i < PREP_TOTAL) {
    int j = i - SZ_E1 - 2 * SZ_SQ - SZ_N1; int n = j >> 7, k = j & 127;
    Wt_n2[j] = (__bf16)W_n2[k * HID + n];
  }
}

// ---------------- kernel 1: zero output accumulators ----------------
__global__ void zero_kernel(float* __restrict__ p, int n) {
  int i = blockIdx.x * blockDim.x + threadIdx.x;
  if (i < n) p[i] = 0.0f;
}

// ---------------- kernel 2: edge pipeline (the hot kernel) ----------------
__global__ void __launch_bounds__(256) edge_kernel(
    const float* __restrict__ h, const float* __restrict__ x,
    const float* __restrict__ ea, const int* __restrict__ eidx,
    const __bf16* __restrict__ Wt_e1, const float* __restrict__ b_e1,
    const __bf16* __restrict__ Wt_e2, const float* __restrict__ b_e2,
    const float* __restrict__ W_g, const float* __restrict__ b_g,
    const __bf16* __restrict__ Wt_x1, const float* __restrict__ b_x1,
    const float* __restrict__ W_x2,
    float* __restrict__ agg, float* __restrict__ dxb) {

  __shared__ __align__(16) __bf16 s_min[64 * IN_EP];   // m_in tile; later aliased by m2
  __shared__ __align__(16) __bf16 s_m1[64 * HID];      // m1 / x-mlp hidden
  __shared__ float s_rel[64][3];
  __shared__ float s_r[64];
  __shared__ float s_g[64];
  __shared__ int   s_dst[64];
  __shared__ int   s_src[64];
  __bf16* s_m2 = s_min;                                // alias (stride HID)

  const int tid = threadIdx.x, lane = tid & 31, wave = tid >> 5;
  const int e0 = blockIdx.x * 64;

  // --- per-edge geometry + RBF + edge_attr ---
  if (tid < 64) {
    const int e = e0 + tid;
    const int si = eidx[e], di = eidx[NEDGE + e];
    s_src[tid] = si; s_dst[tid] = di;
    const float rx = x[3*di+0] - x[3*si+0];
    const float ry = x[3*di+1] - x[3*si+1];
    const float rz = x[3*di+2] - x[3*si+2];
    const float r  = sqrtf(rx*rx + ry*ry + rz*rz + 1e-8f);
    s_rel[tid][0] = rx; s_rel[tid][1] = ry; s_rel[tid][2] = rz; s_r[tid] = r;
    const float step  = CUTOFF / (float)(NUM_G - 1);
    const float coeff = -0.5f / (step * step);
    __bf16* row = s_min + tid * IN_EP;
#pragma unroll
    for (int j = 0; j < NUM_G; ++j) {
      float d = r - step * (float)j;
      row[2 * HID + j] = (__bf16)__expf(coeff * d * d);
    }
#pragma unroll
    for (int j = 0; j < EFEAT; ++j) row[2 * HID + NUM_G + j] = (__bf16)ea[e * EFEAT + j];
#pragma unroll
    for (int j = IN_E; j < IN_EP; ++j) row[j] = (__bf16)0.0f;
  }
  __syncthreads();

  // --- gather h[dst] | h[src] into m_in (coalesced across the 128-wide rows) ---
  for (int i = tid; i < 64 * 256; i += 256) {
    const int e = i >> 8, c = i & 255;
    const int node = (c < HID) ? s_dst[e] : s_src[e];
    s_min[e * IN_EP + c] = (__bf16)h[node * HID + (c & 127)];
  }
  __syncthreads();

  const int mbase = (wave >> 1) * 16;
  const int nb0   = (wave & 1) * 64;
  v8f acc[4];

  // --- GEMM1: m1 = silu(m_in @ W_e1 + b_e1), K=288 ---
  wmma_gemm(s_min, IN_EP, Wt_e1, IN_EP, IN_EP / 32, lane, mbase, nb0, acc);
  store_silu(acc, b_e1, s_m1, HID, lane, mbase, nb0);
  __syncthreads();

  // --- GEMM2: m2 = silu(m1 @ W_e2 + b_e2), K=128 (writes alias of m_in) ---
  wmma_gemm(s_m1, HID, Wt_e2, HID, HID / 32, lane, mbase, nb0, acc);
  store_silu(acc, b_e2, s_m2, HID, lane, mbase, nb0);
  __syncthreads();

  // --- gate g = sigmoid(m2 @ W_g + b_g) (64 edge threads) ---
  if (tid < 64) {
    float gs = b_g[0];
    for (int c = 0; c < HID; ++c) gs += (float)s_m2[tid * HID + c] * W_g[c];
    s_g[tid] = sigmoidf_(gs);
  }
  // --- GEMM3: t = silu(m2 @ W_x1 + b_x1) -> s_m1 (all waves) ---
  wmma_gemm(s_m2, HID, Wt_x1, HID, HID / 32, lane, mbase, nb0, acc);
  store_silu(acc, b_x1, s_m1, HID, lane, mbase, nb0);
  __syncthreads();

  // --- msg scatter: agg[dst] += m2 * g (fp32 global atomics, L2-resident) ---
  for (int i = tid; i < 64 * HID; i += 256) {
    const int e = i >> 7, c = i & 127;
    const float m = (float)s_m2[e * HID + c] * s_g[e];
    atomicAdd(&agg[s_dst[e] * HID + c], m);
  }
  // --- coef = tanh(t @ W_x2); dx[dst] += rel/(r+1)*coef ---
  if (tid < 64) {
    float cs = 0.0f;
    for (int c = 0; c < HID; ++c) cs += (float)s_m1[tid * HID + c] * W_x2[c];
    const float sc = tanhf(cs) / (s_r[tid] + 1.0f);
    const int di = s_dst[tid];
    atomicAdd(&dxb[3 * di + 0], s_rel[tid][0] * sc);
    atomicAdd(&dxb[3 * di + 1], s_rel[tid][1] * sc);
    atomicAdd(&dxb[3 * di + 2], s_rel[tid][2] * sc);
  }
}

// ---------------- kernel 3: node MLP (h_out written over agg, in place) ----------------
__global__ void __launch_bounds__(256) node_kernel(
    const float* __restrict__ h,
    const __bf16* __restrict__ Wt_n1, const float* __restrict__ b_n1,
    const __bf16* __restrict__ Wt_n2, const float* __restrict__ b_n2,
    float* __restrict__ hbuf /* in: agg, out: h_out */) {

  __shared__ __align__(16) __bf16 s_a[64 * 256];
  __shared__ __align__(16) __bf16 s_t[64 * HID];
  const int tid = threadIdx.x, lane = tid & 31, wave = tid >> 5;
  const int n0 = blockIdx.x * 64;

  for (int i = tid; i < 64 * 256; i += 256) {
    const int row = i >> 8, c = i & 255;
    const int node = n0 + row;
    float v = 0.0f;
    if (node < NNODE) v = (c < HID) ? hbuf[node * HID + c] : h[node * HID + (c - HID)];
    s_a[i] = (__bf16)v;
  }
  __syncthreads();

  const int mbase = (wave >> 1) * 16;
  const int nb0   = (wave & 1) * 64;
  v8f acc[4];

  // nm = silu([agg,h] @ W_n1 + b_n1), K=256
  wmma_gemm(s_a, 256, Wt_n1, 256, 256 / 32, lane, mbase, nb0, acc);
  store_silu(acc, b_n1, s_t, HID, lane, mbase, nb0);
  __syncthreads();

  // h_out = h + nm @ W_n2 + b_n2, K=128
  wmma_gemm(s_t, HID, Wt_n2, HID, HID / 32, lane, mbase, nb0, acc);
#pragma unroll
  for (int t = 0; t < 4; ++t) {
    const int n = nb0 + 16 * t + (lane & 15);
    const float b = b_n2[n];
#pragma unroll
    for (int r = 0; r < 8; ++r) {
      const int row = mbase + r + 8 * (lane >> 4);
      const int node = n0 + row;
      if (node < NNODE) hbuf[node * HID + n] = h[node * HID + n] + acc[t][r] + b;
    }
  }
}

// ---------------- kernel 4: x_out = x + dx * mask (in place over dx) ----------------
__global__ void coord_kernel(const float* __restrict__ x, const int* __restrict__ mask,
                             float* __restrict__ xo, int n3) {
  int i = blockIdx.x * blockDim.x + threadIdx.x;
  if (i < n3) {
    const int node = i / 3;
    xo[i] = x[i] + xo[i] * (float)mask[node];
  }
}

// ---------------- launcher ----------------
extern "C" void kernel_launch(void* const* d_in, const int* in_sizes, int n_in,
                              void* d_out, int out_size, void* d_ws, size_t ws_size,
                              hipStream_t stream) {
  const float* h    = (const float*)d_in[0];
  const float* x    = (const float*)d_in[1];
  const float* ea   = (const float*)d_in[2];
  const int*   eidx = (const int*)d_in[3];
  const int*   mask = (const int*)d_in[4];
  const float* W_e1 = (const float*)d_in[5];  const float* b_e1 = (const float*)d_in[6];
  const float* W_e2 = (const float*)d_in[7];  const float* b_e2 = (const float*)d_in[8];
  const float* W_g  = (const float*)d_in[9];  const float* b_g  = (const float*)d_in[10];
  const float* W_n1 = (const float*)d_in[11]; const float* b_n1 = (const float*)d_in[12];
  const float* W_n2 = (const float*)d_in[13]; const float* b_n2 = (const float*)d_in[14];
  const float* W_x1 = (const float*)d_in[15]; const float* b_x1 = (const float*)d_in[16];
  const float* W_x2 = (const float*)d_in[17];

  char* ws = (char*)d_ws;
  __bf16* Wt_e1 = (__bf16*)(ws);
  __bf16* Wt_e2 = (__bf16*)(ws + 2 * SZ_E1);
  __bf16* Wt_x1 = (__bf16*)(ws + 2 * (SZ_E1 + SZ_SQ));
  __bf16* Wt_n1 = (__bf16*)(ws + 2 * (SZ_E1 + 2 * SZ_SQ));
  __bf16* Wt_n2 = (__bf16*)(ws + 2 * (SZ_E1 + 2 * SZ_SQ + SZ_N1));

  float* out = (float*)d_out;
  float* agg = out;                  // N*HID, accumulated then overwritten by h_out
  float* dxb = out + NNODE * HID;    // N*3,  accumulated then overwritten by x_out

  prep_kernel<<<(PREP_TOTAL + 255) / 256, 256, 0, stream>>>(
      W_e1, W_e2, W_x1, W_n1, W_n2, Wt_e1, Wt_e2, Wt_x1, Wt_n1, Wt_n2);

  const int out_total = NNODE * (HID + 3);
  zero_kernel<<<(out_total + 255) / 256, 256, 0, stream>>>(out, out_total);

  edge_kernel<<<NEDGE / 64, 256, 0, stream>>>(
      h, x, ea, eidx, Wt_e1, b_e1, Wt_e2, b_e2, W_g, b_g, Wt_x1, b_x1, W_x2, agg, dxb);

  node_kernel<<<(NNODE + 63) / 64, 256, 0, stream>>>(
      h, Wt_n1, b_n1, Wt_n2, b_n2, agg);

  coord_kernel<<<(NNODE * 3 + 255) / 256, 256, 0, stream>>>(x, mask, dxb, NNODE * 3);
}
